// W2GATFactorExtractor_38010460569737
// MI455X (gfx1250) — matrix operand order repaired and checked by reading, not compiled, not run on hardware
//
#include <hip/hip_runtime.h>
#include <hip/hip_bf16.h>
#include <math.h>

typedef _Float16 half_t;
typedef __attribute__((ext_vector_type(16))) _Float16 v16h;
typedef __attribute__((ext_vector_type(8)))  float    v8f;

#define B_  16
#define T_  32
#define N_  1024
#define D_  64
#define H_  64
#define HEADS_ 4
#define F_  32
#define GH_ 128
#define R_  128
#define O_  32
#define BN_ (B_*N_)   /* 16384 */

__device__ __forceinline__ int lane32() { return threadIdx.x & 31; }

__device__ __forceinline__ v8f zero8() {
  v8f c;
#pragma unroll
  for (int i = 0; i < 8; ++i) c[i] = 0.0f;
  return c;
}

__device__ __forceinline__ v8f wmma_f16(v16h a, v16h b, v8f c) {
  // D = A(16x32 f16) * B(32x16 f16) + C(16x16 f32)
  return __builtin_amdgcn_wmma_f32_16x16x32_f16(false, a, false, b, (short)0, c,
                                                false, false);
}

// ---- A fragment loader: A is row-major with row stride sM, K contiguous. ----
// 16-bit A 16x32 layout: lanes 0-15 row M=lane {K=e + 8*(e>=8)};
// lanes 16-31 row M=lane-16, K shifted by +8.
template<typename AT>
__device__ __forceinline__ v16h load_a(const AT* A, long sM, int row0, int k0) {
  const int l = lane32();
  const AT* p = A + (long)(row0 + (l & 15)) * sM + (k0 + ((l >> 4) << 3));
  v16h a;
#pragma unroll
  for (int e = 0; e < 16; ++e) a[e] = (half_t)p[e + ((e >> 3) << 3)];
  return a;
}

// ---- B fragment loader: element (k,n) at B[k*sK + n*sN]. ----
// 16-bit B 32x16 layout: col = lane&15, K = k0 + e + 8*(e>=8) + 8*(lane>=16).
template<typename BT>
__device__ __forceinline__ v16h load_b(const BT* Bp, long sK, long sN, int k0, int n0) {
  const int l = lane32();
  const long nOff = (long)(n0 + (l & 15)) * sN;
  const int kb = k0 + ((l >> 4) << 3);
  v16h b;
#pragma unroll
  for (int e = 0; e < 16; ++e) b[e] = (half_t)Bp[(long)(kb + e + ((e >> 3) << 3)) * sK + nOff];
  return b;
}

// =====================================================================
// GRU layer: each wave owns a 16-row tile of BN for all T steps.
// Weights staged f32->f16 in LDS; h kept in a wave-private LDS tile so the
// C-layout -> A-fragment transpose between steps goes through DS ops.
// XT = float (layer 0, raw [BN,T,64] view of seq) or half_t (layer 1).
// store_all=1: write [BN,T,64] f16; store_all=0: write only t=T-1 to [BN,64].
// =====================================================================
template<typename XT>
__global__ void __launch_bounds__(128)
gru_kernel(const XT* __restrict__ X, const float* __restrict__ Wih,
           const float* __restrict__ Whh, const float* __restrict__ bih,
           const float* __restrict__ bhh, half_t* __restrict__ Hout,
           int store_all) {
  __shared__ half_t hl[4 * 16 * 64];      // per-wave h tile (f16)
  __shared__ half_t wih_s[192 * 64];      // Wih as [n][k] f16
  __shared__ half_t whh_s[192 * 64];      // Whh as [n][k] f16
  for (int i = threadIdx.x; i < 192 * 64; i += 128) {
    wih_s[i] = (half_t)Wih[i];
    whh_s[i] = (half_t)Whh[i];
  }
  for (int i = threadIdx.x; i < 4 * 16 * 64; i += 128) hl[i] = (half_t)0.0f;
  __syncthreads();

  const int wave = threadIdx.x >> 5;
  const int l = lane32();
  half_t* hw = hl + wave * (16 * 64);
  const long rowbase = ((long)blockIdx.x * 4 + wave) * 16;
  const int col_l = l & 15;
  const int rb = (l >> 4) << 3;

  for (int t = 0; t < T_; ++t) {
    const XT* xrow = X + rowbase * (long)(T_ * 64) + t * 64;
    v16h ax0 = load_a<XT>(xrow, T_ * 64, 0, 0);
    v16h ax1 = load_a<XT>(xrow, T_ * 64, 0, 32);
    v16h ah0 = load_a<half_t>(hw, 64, 0, 0);
    v16h ah1 = load_a<half_t>(hw, 64, 0, 32);
#pragma unroll
    for (int j = 0; j < 4; ++j) {
      const int nr = j * 16, nz = 64 + j * 16, nn0 = 128 + j * 16;
      // gi = x @ Wih^T  => B[k][n] = Wih[n*64+k] : sK=1, sN=64
      v8f cir = zero8(), ciz = zero8(), cin = zero8();
      v8f chr_ = zero8(), chz = zero8(), chn = zero8();
      cir = wmma_f16(ax0, load_b<half_t>(wih_s, 1, 64, 0, nr), cir);
      cir = wmma_f16(ax1, load_b<half_t>(wih_s, 1, 64, 32, nr), cir);
      ciz = wmma_f16(ax0, load_b<half_t>(wih_s, 1, 64, 0, nz), ciz);
      ciz = wmma_f16(ax1, load_b<half_t>(wih_s, 1, 64, 32, nz), ciz);
      cin = wmma_f16(ax0, load_b<half_t>(wih_s, 1, 64, 0, nn0), cin);
      cin = wmma_f16(ax1, load_b<half_t>(wih_s, 1, 64, 32, nn0), cin);
      chr_ = wmma_f16(ah0, load_b<half_t>(whh_s, 1, 64, 0, nr), chr_);
      chr_ = wmma_f16(ah1, load_b<half_t>(whh_s, 1, 64, 32, nr), chr_);
      chz = wmma_f16(ah0, load_b<half_t>(whh_s, 1, 64, 0, nz), chz);
      chz = wmma_f16(ah1, load_b<half_t>(whh_s, 1, 64, 32, nz), chz);
      chn = wmma_f16(ah0, load_b<half_t>(whh_s, 1, 64, 0, nn0), chn);
      chn = wmma_f16(ah1, load_b<half_t>(whh_s, 1, 64, 32, nn0), chn);

      const int col = j * 16 + col_l;
      const float bir = bih[col], biz = bih[64 + col], bin = bih[128 + col];
      const float bhr = bhh[col], bhz = bhh[64 + col], bhn = bhh[128 + col];
#pragma unroll
      for (int r = 0; r < 8; ++r) {
        float rg = 1.0f / (1.0f + expf(-(cir[r] + bir + chr_[r] + bhr)));
        float zg = 1.0f / (1.0f + expf(-(ciz[r] + biz + chz[r] + bhz)));
        float ng = tanhf(cin[r] + bin + rg * (chn[r] + bhn));
        const int hidx = (rb + r) * 64 + col;
        float hp = (float)hw[hidx];
        float hv = (1.0f - zg) * ng + zg * hp;
        hw[hidx] = (half_t)hv;
        if (store_all) {
          Hout[(rowbase + rb + r) * (long)(T_ * 64) + t * 64 + col] = (half_t)hv;
        } else if (t == T_ - 1) {
          Hout[(rowbase + rb + r) * 64L + col] = (half_t)hv;
        }
      }
    }
  }
}

// =====================================================================
// Generic WMMA GEMM: C[M,N] = act(A[M,K] @ B[K,N] + bias (+ skip)).
// One 16x16 output tile per wave; A row-major (K contiguous), B strided.
// =====================================================================
template<typename AT>
__global__ void __launch_bounds__(128)
gemm_kernel(const AT* __restrict__ A, long AsM,
            const float* __restrict__ Bm, long BsK, long BsN,
            const float* __restrict__ bias,
            const float* __restrict__ skip, long skipsM,
            float* __restrict__ C, long CsM,
            int Ntiles, int K, int act) {
  const int tid = blockIdx.x * 4 + (threadIdx.x >> 5);
  const int row0 = (tid / Ntiles) * 16;
  const int n0 = (tid % Ntiles) * 16;
  v8f acc = zero8();
  for (int k0 = 0; k0 < K; k0 += 32) {
    v16h a = load_a<AT>(A, AsM, row0, k0);
    v16h b = load_b<float>(Bm, BsK, BsN, k0, n0);
    acc = wmma_f16(a, b, acc);
  }
  const int l = lane32();
  const int col = n0 + (l & 15);
  const int rb = (l >> 4) << 3;
  const float bv = bias ? bias[col] : 0.0f;
#pragma unroll
  for (int r = 0; r < 8; ++r) {
    float v = acc[r] + bv;
    if (skip) v += skip[(long)(row0 + rb + r) * skipsM + col];
    if (act == 1) v = fmaxf(v, 0.0f);
    C[(long)(row0 + rb + r) * CsM + col] = v;
  }
}

// src/dst attention projections: one thread per (bn, head).
__global__ void srcdst_kernel(const float* __restrict__ Wh,
                              const float* __restrict__ a_src,
                              const float* __restrict__ a_dst,
                              float* __restrict__ srcv, float* __restrict__ dstv) {
  const int idx = blockIdx.x * blockDim.x + threadIdx.x;
  if (idx >= BN_ * HEADS_) return;
  const int bn = idx >> 2, h = idx & 3;
  const float* w = Wh + (long)bn * GH_ + h * F_;
  float s = 0.0f, d = 0.0f;
#pragma unroll
  for (int f = 0; f < F_; ++f) {
    s += w[f] * a_src[h * F_ + f];
    d += w[f] * a_dst[h * F_ + f];
  }
  srcv[idx] = s;
  dstv[idx] = d;
}

// =====================================================================
// GAT: block = (b, 16-row i-block, head). Phase 1: masked leaky-ReLU
// softmax rows (wave32 shfl reductions), write attn f32 to d_out and f16 to
// LDS. Phase 2: 16x32 x K=1024 WMMA GEMM attn@Wh from LDS, K split across
// 2 wave pairs (512 each) with LDS reduction, ELU epilogue.
// =====================================================================
__global__ void __launch_bounds__(128)
gat_kernel(const float* __restrict__ srcv, const float* __restrict__ dstv,
           const int* __restrict__ adj, const float* __restrict__ Wh,
           float* __restrict__ attn_out, float* __restrict__ gat_out) {
  __shared__ half_t attn_s[16 * 1024];  // 32 KB
  __shared__ float red[4 * 256];
  const int b = blockIdx.x >> 6;
  const int ib = blockIdx.x & 63;
  const int h = blockIdx.y;
  const int wave = threadIdx.x >> 5;
  const int l = lane32();

  for (int rr = 0; rr < 4; ++rr) {
    const int il = wave * 4 + rr;
    const int i = ib * 16 + il;
    const float si = srcv[((long)b * N_ + i) * HEADS_ + h];
    float ev[32];
    float m = -3.0e38f;
#pragma unroll
    for (int jj = 0; jj < 32; ++jj) {
      const int j = jj * 32 + l;
      float e = si + dstv[((long)b * N_ + j) * HEADS_ + h];
      e = (e > 0.0f) ? e : 0.2f * e;               // leaky relu
      if (adj[i * N_ + j] == 0) e = -1.0e9f;       // mask
      ev[jj] = e;
      m = fmaxf(m, e);
    }
#pragma unroll
    for (int off = 16; off > 0; off >>= 1) m = fmaxf(m, __shfl_xor(m, off, 32));
    float s = 0.0f;
#pragma unroll
    for (int jj = 0; jj < 32; ++jj) s += expf(ev[jj] - m);
#pragma unroll
    for (int off = 16; off > 0; off >>= 1) s += __shfl_xor(s, off, 32);
    const float inv = 1.0f / s;
#pragma unroll
    for (int jj = 0; jj < 32; ++jj) {
      const int j = jj * 32 + l;
      const float a = expf(ev[jj] - m) * inv;
      attn_out[(((long)b * N_ + i) * N_ + j) * HEADS_ + h] = a;
      attn_s[il * 1024 + j] = (half_t)a;
    }
  }
  __syncthreads();

  // gat[16, 32] = attn_s[16,1024] @ Wh[b,:,h,:] (1024x32)
  const int colt = wave & 1;   // which 16-col tile of F=32
  const int kh = wave >> 1;    // which K half (512)
  const float* Whb = Wh + ((long)b * N_) * GH_ + h * F_;
  v8f acc = zero8();
  for (int kt = 0; kt < 16; ++kt) {
    const int k0 = kh * 512 + kt * 32;
    v16h a = load_a<half_t>(attn_s, 1024, 0, k0);
    v16h bf = load_b<float>(Whb, GH_, 1, k0, colt * 16);
    acc = wmma_f16(a, bf, acc);
  }
#pragma unroll
  for (int r = 0; r < 8; ++r) red[wave * 256 + l * 8 + r] = acc[r];
  __syncthreads();
  if (wave < 2) {
    const int col = h * F_ + colt * 16 + (l & 15);
    const int rb = (l >> 4) << 3;
#pragma unroll
    for (int r = 0; r < 8; ++r) {
      float v = acc[r] + red[(wave + 2) * 256 + l * 8 + r];
      v = (v > 0.0f) ? v : (expf(v) - 1.0f);       // ELU
      gat_out[((long)b * N_ + ib * 16 + rb + r) * GH_ + col] = v;
    }
  }
}

// LayerNorm(32) + tanh: one wave per row, lane = column.
__global__ void __launch_bounds__(128)
ln_kernel(const float* __restrict__ y, const float* __restrict__ g,
          const float* __restrict__ bb, float* __restrict__ out) {
  const int wave = threadIdx.x >> 5;
  const int l = lane32();
  const long row = (long)blockIdx.x * 4 + wave;
  const float v = y[row * O_ + l];
  float mu = v;
#pragma unroll
  for (int off = 16; off > 0; off >>= 1) mu += __shfl_xor(mu, off, 32);
  mu *= (1.0f / 32.0f);
  const float d = v - mu;
  float var = d * d;
#pragma unroll
  for (int off = 16; off > 0; off >>= 1) var += __shfl_xor(var, off, 32);
  var *= (1.0f / 32.0f);
  out[row * O_ + l] = tanhf(d * rsqrtf(var + 1e-5f) * g[l] + bb[l]);
}

extern "C" void kernel_launch(void* const* d_in, const int* in_sizes, int n_in,
                              void* d_out, int out_size, void* d_ws, size_t ws_size,
                              hipStream_t stream) {
  (void)in_sizes; (void)n_in; (void)out_size; (void)ws_size;
  const float* seq  = (const float*)d_in[0];
  const int*   adj  = (const int*)d_in[1];
  const float* Wih0 = (const float*)d_in[2];
  const float* Whh0 = (const float*)d_in[3];
  const float* bih0 = (const float*)d_in[4];
  const float* bhh0 = (const float*)d_in[5];
  const float* Wih1 = (const float*)d_in[6];
  const float* Whh1 = (const float*)d_in[7];
  const float* bih1 = (const float*)d_in[8];
  const float* bhh1 = (const float*)d_in[9];
  const float* gatW = (const float*)d_in[10];
  const float* a_src = (const float*)d_in[11];
  const float* a_dst = (const float*)d_in[12];
  const float* rW1 = (const float*)d_in[13];
  const float* rb1 = (const float*)d_in[14];
  const float* rW2 = (const float*)d_in[15];
  const float* rb2 = (const float*)d_in[16];
  const float* fW  = (const float*)d_in[17];
  const float* fb  = (const float*)d_in[18];
  const float* lng = (const float*)d_in[19];
  const float* lnb = (const float*)d_in[20];
  const float* nW1 = (const float*)d_in[21];
  const float* nb1 = (const float*)d_in[22];
  const float* nW2 = (const float*)d_in[23];
  const float* nb2 = (const float*)d_in[24];

  // Workspace layout (~105 MB total)
  half_t* h1  = (half_t*)d_ws;                       // [BN,T,64] f16
  half_t* h2l = h1 + (size_t)BN_ * T_ * 64;          // [BN,64] f16 (t=T-1 only)
  float* Wh   = (float*)(h2l + (size_t)BN_ * 64);    // [BN,128]
  float* gat  = Wh  + (size_t)BN_ * GH_;             // [BN,128]
  float* r1   = gat + (size_t)BN_ * GH_;             // [BN,128]
  float* res  = r1  + (size_t)BN_ * GH_;             // [BN,128]
  float* y    = res + (size_t)BN_ * GH_;             // [BN,32]
  float* yln  = y   + (size_t)BN_ * O_;              // [BN,32]
  float* t1   = yln + (size_t)BN_ * O_;              // [BN,32]
  float* srcv = t1  + (size_t)BN_ * O_;              // [BN,4]
  float* dstv = srcv + (size_t)BN_ * HEADS_;         // [BN,4]

  float* outp  = (float*)d_out;                      // [B,N,O]
  float* attnp = outp + (size_t)BN_ * O_;            // [B,N,N,HEADS]

  // GRU layer 0 (input f32 raw view [BN,T,64]) and layer 1 (f16 input).
  gru_kernel<float><<<BN_ / 64, 128, 0, stream>>>(seq, Wih0, Whh0, bih0, bhh0, h1, 1);
  gru_kernel<half_t><<<BN_ / 64, 128, 0, stream>>>(h1, Wih1, Whh1, bih1, bhh1, h2l, 0);

  // Wh = gru_feat @ gatW  ([16384x64] x [64x128])
  gemm_kernel<half_t><<<(BN_ / 16) * (GH_ / 16) / 4, 128, 0, stream>>>(
      h2l, 64, gatW, GH_, 1, nullptr, nullptr, 0, Wh, GH_, GH_ / 16, 64, 0);

  srcdst_kernel<<<(BN_ * HEADS_) / 256, 256, 0, stream>>>(Wh, a_src, a_dst, srcv, dstv);

  gat_kernel<<<dim3(B_ * (N_ / 16), HEADS_), 128, 0, stream>>>(
      srcv, dstv, adj, Wh, attnp, gat);

  // Res-C: r1 = relu(gat @ rW1 + rb1); res = relu(gat + r1 @ rW2 + rb2)
  gemm_kernel<float><<<(BN_ / 16) * (R_ / 16) / 4, 128, 0, stream>>>(
      gat, GH_, rW1, R_, 1, rb1, nullptr, 0, r1, R_, R_ / 16, GH_, 1);
  gemm_kernel<float><<<(BN_ / 16) * (R_ / 16) / 4, 128, 0, stream>>>(
      r1, R_, rW2, R_, 1, rb2, gat, GH_, res, R_, R_ / 16, R_, 1);

  // Full-C: y = res @ fW + fb, then LayerNorm + tanh
  gemm_kernel<float><<<(BN_ / 16) * (O_ / 16) / 4, 128, 0, stream>>>(
      res, R_, fW, O_, 1, fb, nullptr, 0, y, O_, O_ / 16, R_, 0);
  ln_kernel<<<BN_ / 4, 128, 0, stream>>>(y, lng, lnb, yln);

  // NN: out = relu(yln @ nW1 + nb1) @ nW2 + nb2  (final writes d_out directly)
  gemm_kernel<float><<<(BN_ / 16) * (O_ / 16) / 4, 128, 0, stream>>>(
      yln, O_, nW1, O_, 1, nb1, nullptr, 0, t1, O_, O_ / 16, O_, 1);
  gemm_kernel<float><<<(BN_ / 16) * (O_ / 16) / 4, 128, 0, stream>>>(
      t1, O_, nW2, O_, 1, nb2, nullptr, 0, outp, O_, O_ / 16, O_, 0);
}
